// Attention_14929306321432
// MI455X (gfx1250) — matrix operand order repaired and checked by reading, not compiled, not run on hardware
//
#include <hip/hip_runtime.h>
#include <hip/hip_bf16.h>

typedef __attribute__((ext_vector_type(16))) _Float16 v16h;
typedef __attribute__((ext_vector_type(8)))  _Float16 v8h;
typedef __attribute__((ext_vector_type(8)))  float    v8f;
typedef __attribute__((ext_vector_type(4)))  int      v4i;

typedef __attribute__((address_space(1))) v4i* gas_v4i;   // global (AS1)
typedef __attribute__((address_space(3))) v4i* las_v4i;   // LDS (AS3)

#define B_   4
#define L_   2048
#define D_   1024
#define H_   16
#define DH_  64
#define NEGV 1e7f

#if defined(__has_builtin)
#  if __has_builtin(__builtin_amdgcn_global_load_async_to_lds_b128)
#    define HAVE_ASYNC_LDS 1
#  endif
#endif

__device__ __forceinline__ v16h mk16(v8h lo, v8h hi) {
  v16h r;
  #pragma unroll
  for (int i = 0; i < 8; ++i) { r[i] = lo[i]; r[i + 8] = hi[i]; }
  return r;
}

// Copy 16 bytes global -> LDS. Async (ASYNCcnt-tracked) when available.
__device__ __forceinline__ void stage16(const _Float16* gsrc, _Float16* ldst) {
#ifdef HAVE_ASYNC_LDS
  __builtin_amdgcn_global_load_async_to_lds_b128(
      (gas_v4i)(_Float16*)gsrc, (las_v4i)ldst, 0, 0);
#else
  *(v8h*)ldst = *(const v8h*)gsrc;
#endif
}

__device__ __forceinline__ void async_wait0() {
#ifdef HAVE_ASYNC_LDS
#  if __has_builtin(__builtin_amdgcn_s_wait_asynccnt)
  __builtin_amdgcn_s_wait_asynccnt(0);
#  else
  asm volatile("s_wait_asynccnt 0" ::: "memory");
#  endif
#endif
}

// ---------------------------------------------------------------------------
// Weight convert: fp32 W[k,n] (row-major DxD) -> f16 Wt[n,k] (row-major DxD)
// ---------------------------------------------------------------------------
__global__ void wconv_kernel(const float* __restrict__ W, _Float16* __restrict__ Wt) {
  int idx = blockIdx.x * blockDim.x + threadIdx.x;   // D*D threads
  int k = idx >> 10;
  int n = idx & (D_ - 1);
  Wt[(size_t)n * D_ + k] = (_Float16)W[idx];
}

// ---------------------------------------------------------------------------
// x (+ positional encoding) -> f16
// ---------------------------------------------------------------------------
__global__ void addpe_kernel(const float* __restrict__ x, _Float16* __restrict__ out, int use_pe) {
  size_t idx = (size_t)blockIdx.x * blockDim.x + threadIdx.x;  // B*L*D threads
  int d = (int)(idx & (D_ - 1));
  int l = (int)((idx >> 10) & (L_ - 1));
  float v = x[idx];
  if (use_pe) {
    int p = d >> 1;
    // periods = 10000^(-p/512) = exp(-(p/512)*ln(10000))
    float period = __expf(-((float)p * (1.0f / 512.0f)) * 9.210340371976184f);
    float ang = (float)l * period;
    v += (d & 1) ? __cosf(ang) : __sinf(ang);
  }
  out[idx] = (_Float16)v;
}

// ---------------------------------------------------------------------------
// GEMM: C[m,n] = sum_k A[m,k] * Wt[n,k] + bias[n]
//   A:  f16 [M, K] row-major      Wt: f16 [N=1024, K] row-major
//   8 waves / block; wave computes a 16x64 tile. Grid: (M/128, 1024/64).
//   Wt k-slab (64 cols x 32 k = 4KB) staged once per block into LDS with
//   double-buffered async global->LDS copies; k-loop unrolled 2x so the
//   ping-pong buffer index is a compile-time constant.
//   mode 0: store f16 head-split [B,H,L,64]
//   mode 2: store f16 head-split transposed [B,H,64,L]
//   mode 3: store fp32 [M, 1024]
// ---------------------------------------------------------------------------
__global__ void gemm_f16_kernel(const _Float16* __restrict__ A,
                                const _Float16* __restrict__ Wt,
                                const float*    __restrict__ bias,
                                void* __restrict__ outp,
                                int M, int K, int mode) {
  __shared__ __align__(16) _Float16 sB[2][64 * 32];   // 2 x 4KB
  const int tid  = threadIdx.x;
  const int lane = tid & 31;
  const int wave = tid >> 5;
  const int hi   = lane >> 4;       // half-wave select
  const int ln   = lane & 15;
  const int m0   = blockIdx.x * 128 + wave * 16;
  const int n0   = blockIdx.y * 64;

  // staging role: thread copies 8 halves of column (n0+c), chunk 'part'
  const int c    = tid >> 2;        // 0..63
  const int part = tid & 3;         // 0..3
  const _Float16* gB = Wt + (size_t)(n0 + c) * K + part * 8;
  _Float16* lB0 = &sB[0][c * 32 + part * 8];
  _Float16* lB1 = &sB[1][c * 32 + part * 8];

  v8f acc[4];
  #pragma unroll
  for (int t = 0; t < 4; ++t) {
    float bv = bias[n0 + t * 16 + ln];
    #pragma unroll
    for (int r = 0; r < 8; ++r) acc[t][r] = bv;
  }

  const _Float16* aRow = A + (size_t)(m0 + ln) * K;

  // one k-step: A frag from global, 4 B frags from LDS slab, 4 WMMAs
  auto step = [&](const _Float16* slab, int kb) {
    v8h al = *(const v8h*)(aRow + kb + 16 * hi);
    v8h ah = *(const v8h*)(aRow + kb + 16 * hi + 8);
    v16h a = mk16(al, ah);
    v16h b0, b1, b2, b3;
    {
      const _Float16* p0 = slab + (0 * 16 + ln) * 32 + 16 * hi;
      const _Float16* p1 = slab + (1 * 16 + ln) * 32 + 16 * hi;
      const _Float16* p2 = slab + (2 * 16 + ln) * 32 + 16 * hi;
      const _Float16* p3 = slab + (3 * 16 + ln) * 32 + 16 * hi;
      b0 = mk16(*(const v8h*)p0, *(const v8h*)(p0 + 8));
      b1 = mk16(*(const v8h*)p1, *(const v8h*)(p1 + 8));
      b2 = mk16(*(const v8h*)p2, *(const v8h*)(p2 + 8));
      b3 = mk16(*(const v8h*)p3, *(const v8h*)(p3 + 8));
    }
    acc[0] = __builtin_amdgcn_wmma_f32_16x16x32_f16(false, a, false, b0, (short)0, acc[0], false, false);
    acc[1] = __builtin_amdgcn_wmma_f32_16x16x32_f16(false, a, false, b1, (short)0, acc[1], false, false);
    acc[2] = __builtin_amdgcn_wmma_f32_16x16x32_f16(false, a, false, b2, (short)0, acc[2], false, false);
    acc[3] = __builtin_amdgcn_wmma_f32_16x16x32_f16(false, a, false, b3, (short)0, acc[3], false, false);
  };

  stage16(gB, lB0);                        // prologue: k-slab 0 -> buffer 0
  for (int kb = 0; kb < K; kb += 64) {
    async_wait0();
    __syncthreads();                       // buffer 0 resident
    if (kb + 32 < K) stage16(gB + kb + 32, lB1);
    step(sB[0], kb);

    async_wait0();
    __syncthreads();                       // buffer 1 resident
    if (kb + 64 < K) stage16(gB + kb + 64, lB0);
    step(sB[1], kb + 32);
  }

  // C layout: c[r] -> row m0 + r + 8*hi, col n0 + t*16 + ln
  #pragma unroll
  for (int t = 0; t < 4; ++t) {
    #pragma unroll
    for (int r = 0; r < 8; ++r) {
      int m = m0 + r + 8 * hi;
      int n = n0 + t * 16 + ln;
      float v = acc[t][r];
      if (mode == 3) {
        ((float*)outp)[(size_t)m * D_ + n] = v;
      } else {
        int bb = m >> 11;            // / L_
        int l  = m & (L_ - 1);
        int hh = n >> 6;             // head
        int dc = n & 63;             // within-head dim
        _Float16* o16 = (_Float16*)outp;
        if (mode == 2)
          o16[(((size_t)bb * H_ + hh) * DH_ + dc) * L_ + l] = (_Float16)v;  // Vt
        else
          o16[(((size_t)bb * H_ + hh) * L_ + l) * DH_ + dc] = (_Float16)v;  // Qh/Kh
      }
    }
  }
}

// ---------------------------------------------------------------------------
// Flash attention: causal, online softmax.
//   Qh/Kh f16 [B,H,L,64]; Vt f16 [B,H,64,L]; pad u8 [B,L]; Oh f16 [B,L,D]
//   4 waves / block, each wave owns 16 query rows of a 64-row query block.
// ---------------------------------------------------------------------------
__global__ void flash_kernel(const _Float16* __restrict__ Qh,
                             const _Float16* __restrict__ Kh,
                             const _Float16* __restrict__ Vt,
                             const unsigned char* __restrict__ pad,
                             _Float16* __restrict__ Oh) {
  __shared__ __align__(16) _Float16 smem[4 * 16 * 32];
  const int lane = threadIdx.x & 31;
  const int wave = threadIdx.x >> 5;
  const int hi   = lane >> 4;
  const int ln   = lane & 15;

  int blk  = blockIdx.x;            // B*H*(L/64) blocks
  int qblk = blk % (L_ / 64);
  int bh   = blk / (L_ / 64);
  int h    = bh % H_;
  int b    = bh / H_;
  int q0   = qblk * 64 + wave * 16;

  const _Float16* Qb = Qh + (size_t)bh * L_ * DH_;
  const _Float16* Kb = Kh + (size_t)bh * L_ * DH_;
  const _Float16* Vb = Vt + (size_t)bh * DH_ * L_;
  _Float16* lds = smem + wave * (16 * 32);

  // Q A-fragments for d-chunks 0..31 and 32..63 (loaded once)
  v16h qa[2];
  #pragma unroll
  for (int ks = 0; ks < 2; ++ks) {
    const _Float16* p = Qb + (size_t)(q0 + ln) * DH_ + ks * 32 + 16 * hi;
    qa[ks] = mk16(*(const v8h*)p, *(const v8h*)(p + 8));
  }

  float mrun[8], lrun[8];
  v8f o[4];
  #pragma unroll
  for (int r = 0; r < 8; ++r) { mrun[r] = -3.0e38f; lrun[r] = 0.0f; }
  #pragma unroll
  for (int t = 0; t < 4; ++t)
    #pragma unroll
    for (int r = 0; r < 8; ++r) o[t][r] = 0.0f;

  const float scale = 0.125f;                  // 1/sqrt(64)
  const int kend = q0 + 16;                    // keys needed: <= q0+15 (causal)

  for (int kb = 0; kb < kend; kb += 32) {
    // prefetch next key block's K/V cachelines (global_prefetch_b8)
    if (kb + 32 < kend) {
      __builtin_prefetch(Kb + (size_t)(kb + 32 + ln) * DH_, 0, 0);
      __builtin_prefetch(Vb + (size_t)ln * L_ + kb + 32, 0, 0);
    }

    // ---- scores: S = Q K^T for two 16-key tiles (load all 4 frags first) ----
    v16h kf[2][2];
    #pragma unroll
    for (int t = 0; t < 2; ++t)
      #pragma unroll
      for (int ks = 0; ks < 2; ++ks) {
        const _Float16* p = Kb + (size_t)(kb + t * 16 + ln) * DH_ + ks * 32 + 16 * hi;
        kf[t][ks] = mk16(*(const v8h*)p, *(const v8h*)(p + 8));
      }
    v8f s[2];
    #pragma unroll
    for (int t = 0; t < 2; ++t) {
      #pragma unroll
      for (int r = 0; r < 8; ++r) s[t][r] = 0.0f;
      s[t] = __builtin_amdgcn_wmma_f32_16x16x32_f16(
          false, qa[0], false, kf[t][0], (short)0, s[t], false, false);
      s[t] = __builtin_amdgcn_wmma_f32_16x16x32_f16(
          false, qa[1], false, kf[t][1], (short)0, s[t], false, false);
    }

    // ---- scale + causal/padding mask ----
    float sv[2][8];
    #pragma unroll
    for (int t = 0; t < 2; ++t) {
      int key = kb + t * 16 + ln;
      float padneg = pad[(size_t)b * L_ + key] ? NEGV : 0.0f;
      #pragma unroll
      for (int r = 0; r < 8; ++r) {
        int qrow = q0 + r + 8 * hi;
        float v = s[t][r] * scale - padneg;
        if (key > qrow) v -= NEGV;
        sv[t][r] = v;
      }
    }

    // ---- online softmax (row = 16 lanes of a half-wave) ----
    #pragma unroll
    for (int r = 0; r < 8; ++r) {
      float tmax = fmaxf(sv[0][r], sv[1][r]);
      #pragma unroll
      for (int off = 1; off < 16; off <<= 1)
        tmax = fmaxf(tmax, __shfl_xor(tmax, off, 32));
      float mnew = fmaxf(mrun[r], tmax);
      float corr = __expf(mrun[r] - mnew);
      float p0 = __expf(sv[0][r] - mnew);
      float p1 = __expf(sv[1][r] - mnew);
      float rs = p0 + p1;
      #pragma unroll
      for (int off = 1; off < 16; off <<= 1)
        rs += __shfl_xor(rs, off, 32);
      lrun[r] = lrun[r] * corr + rs;
      mrun[r] = mnew;
      #pragma unroll
      for (int t = 0; t < 4; ++t) o[t][r] *= corr;
      // stash P into LDS in plain row-major 16x32 (C-layout -> memory)
      int m = r + 8 * hi;
      lds[m * 32 + ln]      = (_Float16)p0;
      lds[m * 32 + 16 + ln] = (_Float16)p1;
    }
    asm volatile("" ::: "memory");   // keep LDS write->read order in the compiler

    // ---- re-load P as a 16x32 A fragment ----
    const _Float16* pp = lds + ln * 32 + 16 * hi;
    v16h pa = mk16(*(const v8h*)pp, *(const v8h*)(pp + 8));

    // ---- O += P * V (Vt rows are contiguous in key dim) ----
    v16h vf[4];
    #pragma unroll
    for (int t = 0; t < 4; ++t) {
      const _Float16* vp = Vb + (size_t)(t * 16 + ln) * L_ + kb + 16 * hi;
      vf[t] = mk16(*(const v8h*)vp, *(const v8h*)(vp + 8));
    }
    #pragma unroll
    for (int t = 0; t < 4; ++t)
      o[t] = __builtin_amdgcn_wmma_f32_16x16x32_f16(
          false, pa, false, vf[t], (short)0, o[t], false, false);
  }

  // ---- normalize + store merged-head f16 [B,L,D] ----
  #pragma unroll
  for (int r = 0; r < 8; ++r) {
    float inv = 1.0f / lrun[r];
    int q = q0 + r + 8 * hi;
    #pragma unroll
    for (int t = 0; t < 4; ++t) {
      int col = h * DH_ + t * 16 + ln;
      Oh[((size_t)b * L_ + q) * D_ + col] = (_Float16)(o[t][r] * inv);
    }
  }
}

// ---------------------------------------------------------------------------
extern "C" void kernel_launch(void* const* d_in, const int* in_sizes, int n_in,
                              void* d_out, int out_size, void* d_ws, size_t ws_size,
                              hipStream_t stream) {
  const float* q   = (const float*)d_in[0];
  const float* k   = (const float*)d_in[1];
  const float* v   = (const float*)d_in[2];
  const unsigned char* pad = (const unsigned char*)d_in[3];
  const float* Wq  = (const float*)d_in[4];
  const float* bq  = (const float*)d_in[5];
  const float* Wk  = (const float*)d_in[6];
  const float* bk  = (const float*)d_in[7];
  const float* Wv  = (const float*)d_in[8];
  const float* bv  = (const float*)d_in[9];
  const float* Wo  = (const float*)d_in[10];
  const float* bo  = (const float*)d_in[11];
  float* out = (float*)d_out;

  char* ws = (char*)d_ws;
  const size_t SZ = (size_t)B_ * L_ * D_ * sizeof(_Float16);   // 16 MB
  _Float16* qpe = (_Float16*)(ws + 0 * SZ);
  _Float16* kpe = (_Float16*)(ws + 1 * SZ);
  _Float16* vh  = (_Float16*)(ws + 2 * SZ);
  _Float16* Qh  = (_Float16*)(ws + 3 * SZ);
  _Float16* Kh  = (_Float16*)(ws + 4 * SZ);
  _Float16* Vt  = (_Float16*)(ws + 5 * SZ);
  _Float16* Ah  = (_Float16*)(ws + 6 * SZ);
  _Float16* Wqt = (_Float16*)(ws + 7 * SZ);
  _Float16* Wkt = Wqt + (size_t)D_ * D_;
  _Float16* Wvt = Wkt + (size_t)D_ * D_;
  _Float16* Wot = Wvt + (size_t)D_ * D_;

  const int M = B_ * L_;

  // 1) weight transpose+convert
  wconv_kernel<<<(D_ * D_) / 256, 256, 0, stream>>>(Wq, Wqt);
  wconv_kernel<<<(D_ * D_) / 256, 256, 0, stream>>>(Wk, Wkt);
  wconv_kernel<<<(D_ * D_) / 256, 256, 0, stream>>>(Wv, Wvt);
  wconv_kernel<<<(D_ * D_) / 256, 256, 0, stream>>>(Wo, Wot);

  // 2) PE add + convert
  addpe_kernel<<<(M * D_) / 256, 256, 0, stream>>>(q, qpe, 1);
  addpe_kernel<<<(M * D_) / 256, 256, 0, stream>>>(k, kpe, 1);
  addpe_kernel<<<(M * D_) / 256, 256, 0, stream>>>(v, vh, 0);

  // 3) projections (WMMA GEMMs), head-split stores
  dim3 ggrid(M / 128, D_ / 64);
  gemm_f16_kernel<<<ggrid, 256, 0, stream>>>(qpe, Wqt, bq, Qh, M, D_, 0);
  gemm_f16_kernel<<<ggrid, 256, 0, stream>>>(kpe, Wkt, bk, Kh, M, D_, 0);
  gemm_f16_kernel<<<ggrid, 256, 0, stream>>>(vh,  Wvt, bv, Vt, M, D_, 2);

  // 4) causal flash attention
  flash_kernel<<<B_ * H_ * (L_ / 64), 128, 0, stream>>>(Qh, Kh, Vt, pad, Ah);

  // 5) output projection (fp32 out + bias)
  gemm_f16_kernel<<<ggrid, 256, 0, stream>>>(Ah, Wot, bo, out, M, D_, 3);
}
